// SelfAttnWithMask_22376779612347
// MI455X (gfx1250) — matrix operand order repaired and checked by reading, not compiled, not run on hardware
//
#include <hip/hip_runtime.h>
#include <hip/hip_bf16.h>

typedef __attribute__((ext_vector_type(16))) __bf16 v16bf;
typedef __attribute__((ext_vector_type(8)))  float  v8f;
typedef unsigned int u32x4 __attribute__((ext_vector_type(4)));
typedef unsigned int u32x8 __attribute__((ext_vector_type(8)));

#define B_     8
#define S_     2048
#define D_     1024
#define QT     16          // query rows per workgroup
#define KT     32          // keys per step (matches WMMA K=32)
#define NWAVE  8
#define DSL    128         // D columns per wave
#define KSTEPS (S_ / KT)   // 64

union VecA  { v16bf v; unsigned u[8]; };
union Pack2 { __bf16 h[2]; unsigned u; };

// LDS layout (dynamic):
//   [0      , 131072) : kv tile, 32 x 1024 f32 (TDM destination; also Q staging)
//   [131072 , 147456) : score partial slabs, 8 waves x 16 x 32 f32
//   [147456 , 148480) : P tile, 16 x 32 bf16
//   [148480 , 148544) : corr[16] f32
//   [148544 , 148608) : l[16]    f32
#define SMEM_BYTES (131072 + 16384 + 1024 + 64 + 64)

// Issue a TDM DMA of a (rows x 1024) f32 tile from global memory into LDS.
// D# group0: [1:0]=count=1, [63:32]=lds_addr, [120:64]=global_addr, [127:126]=type=2
// D# group1: [17:16]=data_size=2 (4B), tensor_dim0=1024, tensor_dim1=rows,
//            tile_dim0=1024, tile_dim1=rows, tensor_dim0_stride=1024
__device__ __forceinline__ void tdm_load_tile(const float* gsrc, unsigned lds_off, unsigned rows)
{
    unsigned long long ga = (unsigned long long)(uintptr_t)gsrc;
    u32x4 g0;
    g0[0] = 1u;                                    // count=1, user descriptor
    g0[1] = lds_off;                               // LDS byte address
    g0[2] = (unsigned)ga;                          // global addr [31:0]
    g0[3] = (unsigned)(ga >> 32) | (2u << 30);     // global addr [56:32] | type=2
    u32x8 g1;
    g1[0] = 0x00020000u;                           // wg_mask=0, data_size=4B
    g1[1] = (unsigned)D_ << 16;                    // tensor_dim0 lo16 (1024)
    g1[2] = rows << 16;                            // tensor_dim0 hi16=0 | tensor_dim1 lo16
    g1[3] = (unsigned)D_ << 16;                    // tensor_dim1 hi16=0 | tile_dim0 (1024)
    g1[4] = rows;                                  // tile_dim1 | tile_dim2=0
    g1[5] = (unsigned)D_;                          // tensor_dim0_stride lo32 (1024)
    g1[6] = 0u;                                    // stride hi16 | tensor_dim1_stride lo16
    g1[7] = 0u;
    asm volatile("tensor_load_to_lds %0, %1" :: "s"(g0), "s"(g1) : "memory");
}

__global__ __launch_bounds__(256)
void SelfAttnWithMask_22376779612347_kernel(const float* __restrict__ x,
                                            const float* __restrict__ mask,
                                            float* __restrict__ out)
{
    extern __shared__ char smem[];
    float*  kvf  = (float*)smem;                        // [KT][D_] f32
    float*  slab = (float*)(smem + 131072);             // [NWAVE][QT][KT] f32
    __bf16* pbuf = (__bf16*)(smem + 131072 + 16384);    // [QT][KT] bf16
    float*  corr = (float*)(smem + 131072 + 16384 + 1024);
    float*  lbuf = (float*)(smem + 131072 + 16384 + 1024 + 64);

    const int tid  = threadIdx.x;
    const int wave = tid >> 5;
    const int lane = tid & 31;
    const int lrow = lane & 15;     // row (A/C) or col (B) within 16
    const int half = lane >> 4;     // lane half selects K/M group

    const int wg = blockIdx.x;
    const int b  = wg / (S_ / QT);
    const int q0 = (wg % (S_ / QT)) * QT;

    const unsigned kv_lds_off = (unsigned)(uintptr_t)kvf;  // LDS aperture low bits = byte offset

    // ---------- TDM-stage Q tile (16 x 1024 f32) into kv buffer ----------
    if (wave == 0) {
        tdm_load_tile(x + (size_t)(b * S_ + q0) * D_, kv_lds_off, QT);
        __builtin_amdgcn_s_wait_tensorcnt(0);
    }
    __syncthreads();

    // A-operand (16-bit 16x32) K index: kA(v) = (v/4)*16 + half*8 + (v%4)*2
    VecA qa[4];
#pragma unroll
    for (int c = 0; c < 4; ++c) {
#pragma unroll
        for (int v = 0; v < 8; ++v) {
            int kA = ((v >> 2) << 4) + (half << 3) + ((v & 3) << 1);
            int d  = wave * DSL + c * 32 + kA;
            float2 f = *(const float2*)(kvf + lrow * D_ + d);
            Pack2 p; p.h[0] = (__bf16)f.x; p.h[1] = (__bf16)f.y;
            qa[c].u[v] = p.u;
        }
    }

    v8f facc[8];
#pragma unroll
    for (int j = 0; j < 8; ++j) facc[j] = {};

    float m_i = -__builtin_inff();   // meaningful only for tid < 16
    float l_i = 0.0f;

    for (int kk = 0; kk < KSTEPS; ++kk) {
        __syncthreads();  // all waves done reading kv (incl. Q extraction at kk==0)

        // ---------- TDM-stage KV tile (32 keys x 1024 f32) ----------
        if (wave == 0) {
            tdm_load_tile(x + (size_t)(b * S_ + kk * KT) * D_, kv_lds_off, KT);
            __builtin_amdgcn_s_wait_tensorcnt(0);
        }
        if (kk + 1 < KSTEPS)   // global_prefetch_b8 of next tile into L2
            __builtin_prefetch(x + (size_t)(b * S_ + (kk + 1) * KT) * D_ + tid * 128, 0, 1);
        __syncthreads();

        // ---------- partial scores over this wave's 128-wide D slice ----------
        v8f sacc0 = {}, sacc1 = {};
#pragma unroll
        for (int c = 0; c < 4; ++c) {
            VecA bm0, bm1;
#pragma unroll
            for (int v = 0; v < 8; ++v) {
                int kB = ((v >> 2) << 4) + (half << 3) + ((v & 3) << 1);     // B-operand K index
                int d  = wave * DSL + c * 32 + kB;
                float2 f0 = *(const float2*)(kvf + (lrow)      * D_ + d);
                float2 f1 = *(const float2*)(kvf + (16 + lrow) * D_ + d);
                Pack2 p0; p0.h[0] = (__bf16)f0.x; p0.h[1] = (__bf16)f0.y;
                Pack2 p1; p1.h[0] = (__bf16)f1.x; p1.h[1] = (__bf16)f1.y;
                bm0.u[v] = p0.u;
                bm1.u[v] = p1.u;
            }
            sacc0 = __builtin_amdgcn_wmma_f32_16x16x32_bf16(false, qa[c].v, false, bm0.v,
                                                            (short)0, sacc0, false, false);
            sacc1 = __builtin_amdgcn_wmma_f32_16x16x32_bf16(false, qa[c].v, false, bm1.v,
                                                            (short)0, sacc1, false, false);
        }
        // C layout: VGPR r -> row = r + 8*half, col = lrow (+16 for tile 1)
#pragma unroll
        for (int r = 0; r < 8; ++r) {
            int row = r + (half << 3);
            slab[(wave * QT + row) * KT + lrow]      = sacc0[r];
            slab[(wave * QT + row) * KT + 16 + lrow] = sacc1[r];
        }
        __syncthreads();

        // ---------- online softmax (wave 0, lanes 0..15; one row each) ----------
        if (tid < QT) {
            int row = tid;
            float sv[KT];
            float mnew = m_i;
#pragma unroll
            for (int col = 0; col < KT; ++col) {
                float s = 0.0f;
#pragma unroll
                for (int w = 0; w < NWAVE; ++w)      // deterministic serial sum in wave order
                    s += slab[(w * QT + row) * KT + col];
                s += mask[(size_t)b * S_ + kk * KT + col];
                sv[col] = s;
                mnew = fmaxf(mnew, s);
            }
            float cf  = __expf(m_i - mnew);
            float sum = 0.0f;
#pragma unroll
            for (int col = 0; col < KT; ++col) {
                float e = __expf(sv[col] - mnew);
                sum += e;
                pbuf[row * KT + col] = (__bf16)e;
            }
            l_i = l_i * cf + sum;
            m_i = mnew;
            corr[row] = cf;
        }
        __syncthreads();

        // ---------- rescale accumulators, then P (16x32) x V (32x128-slice) ----------
        float cr[8];
#pragma unroll
        for (int r = 0; r < 8; ++r) cr[r] = corr[r + (half << 3)];
#pragma unroll
        for (int j = 0; j < 8; ++j)
#pragma unroll
            for (int r = 0; r < 8; ++r) facc[j][r] *= cr[r];

        VecA pA;
#pragma unroll
        for (int v = 0; v < 8; ++v) {
            int kA = ((v >> 2) << 4) + (half << 3) + ((v & 3) << 1);
            pA.u[v] = *(const unsigned*)(pbuf + lrow * KT + kA);
        }
#pragma unroll
        for (int j = 0; j < 8; ++j) {
            int ncol = wave * DSL + j * 16 + lrow;
            VecA bm;
#pragma unroll
            for (int v = 0; v < 8; ++v) {
                int k0 = ((v >> 2) << 4) + (half << 3) + ((v & 3) << 1);     // two keys per VGPR
                Pack2 pk;
                pk.h[0] = (__bf16)kvf[(size_t)k0       * D_ + ncol];
                pk.h[1] = (__bf16)kvf[(size_t)(k0 + 1) * D_ + ncol];
                bm.u[v] = pk.u;
            }
            facc[j] = __builtin_amdgcn_wmma_f32_16x16x32_bf16(false, pA.v, false, bm.v,
                                                              (short)0, facc[j], false, false);
        }
    }

    // ---------- finalize: divide by l, store ----------
    if (tid < QT) lbuf[tid] = l_i;
    __syncthreads();

    float inv[8];
#pragma unroll
    for (int r = 0; r < 8; ++r) inv[r] = 1.0f / lbuf[r + (half << 3)];

#pragma unroll
    for (int j = 0; j < 8; ++j) {
#pragma unroll
        for (int r = 0; r < 8; ++r) {
            int row = r + (half << 3);
            int col = wave * DSL + j * 16 + lrow;
            out[(size_t)(b * S_ + q0 + row) * D_ + col] = facc[j][r] * inv[r];
        }
    }
}

extern "C" void kernel_launch(void* const* d_in, const int* in_sizes, int n_in,
                              void* d_out, int out_size, void* d_ws, size_t ws_size,
                              hipStream_t stream) {
    (void)in_sizes; (void)n_in; (void)d_ws; (void)ws_size; (void)out_size;
    const float* x    = (const float*)d_in[0];   // encoded_seq [8,2048,1024] f32
    const float* mask = (const float*)d_in[1];   // inf_mask    [8,2048]      f32
    float*       out  = (float*)d_out;           // [8,2048,1024] f32

    dim3 grid(B_ * (S_ / QT));   // 1024 workgroups
    dim3 block(256);             // 8 wave32s
    SelfAttnWithMask_22376779612347_kernel<<<grid, block, SMEM_BYTES, stream>>>(x, mask, out);
}
